// iRadonMap_61881888800894
// MI455X (gfx1250) — matrix operand order repaired and checked by reading, not compiled, not run on hardware
//
#include <hip/hip_runtime.h>
#include <hip/hip_bf16.h>

typedef __attribute__((ext_vector_type(16))) _Float16 v16h;
typedef __attribute__((ext_vector_type(8)))  _Float16 v8h;
typedef __attribute__((ext_vector_type(8)))  float    v8f;
typedef __attribute__((ext_vector_type(4)))  int      v4i;

#define VIEWS 360
#define NDET  736
#define NX    512
#define NY    512
#define MPAD  368          // 360 padded to 16
#define KTILES (NDET/32)   // 23
#define BP_SCALE 0.00872665f  // STEP/2

#if defined(__has_builtin)
#if __has_builtin(__builtin_amdgcn_global_load_async_to_lds_b128)
#define HAS_ASYNC_LDS 1
#endif
#if __has_builtin(__builtin_amdgcn_s_wait_asynccnt)
#define HAS_WAIT_ASYNC 1
#endif
#endif

typedef __attribute__((address_space(1))) v4i glob_v4i;
typedef __attribute__((address_space(3))) v4i lds_v4i;

// ---------------------------------------------------------------------------
// convert f32 -> f16 with optional zero tail padding (n >= nsrc)
__global__ __launch_bounds__(256) void cvt_f16_k(const float* __restrict__ src,
                                                 _Float16* __restrict__ dst,
                                                 int n, int nsrc) {
    int i = blockIdx.x * 256 + threadIdx.x;
    if (i < n) dst[i] = (i < nsrc) ? (_Float16)src[i] : (_Float16)0.0f;
}

// ---------------------------------------------------------------------------
// Pack the 22 resblock conv weights (f32 [64][64][3][3]) into per-lane WMMA
// A-fragment order: frag f = ((co_t*3+ky)*3+kx)*2 + cb, 32 lanes x 16 halves.
// A k order used here (16x32 f16, same as B): lane<16 -> h[0..15]=K0..15;
//                       lane>=16 -> h[0..15]=K16..31.
__global__ __launch_bounds__(32) void pack_conv_w_k(const float* __restrict__ bw1,
                                                    const float* __restrict__ bw2,
                                                    _Float16* __restrict__ wpack) {
    int bid  = blockIdx.x;            // conv*72 + f, conv in [0,22)
    int conv = bid / 72;
    int f    = bid % 72;
    int blk  = conv >> 1;
    const float* w = (conv & 1) ? (bw2 + (size_t)blk * 64 * 64 * 9)
                                : (bw1 + (size_t)blk * 64 * 64 * 9);
    int cb   = f & 1;
    int rest = f >> 1;                // (co_t*3+ky)*3+kx
    int kx   = rest % 3;
    int tmp  = rest / 3;
    int ky   = tmp % 3;
    int co_t = tmp / 3;

    int lane = threadIdx.x;
    int m    = lane & 15;
    int base = (lane >> 4) * 16;
    v16h o;
#pragma unroll
    for (int j = 0; j < 16; ++j) {
        int kl = base + j;  // K within 32-chunk
        int ci = cb * 32 + kl;
        int co = co_t * 16 + m;
        o[j] = (_Float16)w[((co * 64 + ci) * 3 + ky) * 3 + kx];
    }
    *(v16h*)&wpack[((size_t)bid * 32 + lane) * 16] = o;
}

// ---------------------------------------------------------------------------
// Dense layer: D = relu(A(MxK) * B^T + bias), A=[MPAD][736] f16, B=w=[736][736] f16
// (w row n supplies B column n along K -> contiguous loads).
// One wave per 16x16 tile. flags: 1 -> write f16 out, 2 -> write f32 out (v<360).
__global__ __launch_bounds__(32) void gemm_wmma_k(const _Float16* __restrict__ A16,
                                                  const _Float16* __restrict__ W16,
                                                  const float* __restrict__ bias,
                                                  _Float16* __restrict__ outH,
                                                  float* __restrict__ outF,
                                                  int flags) {
    int mt = blockIdx.x;              // 0..22
    int nt = blockIdx.y;              // 0..45
    int lane = threadIdx.x;
    int lm = lane & 15, lh = lane >> 4;
    int vrow = mt * 16 + lm;
    int ncol = nt * 16 + lm;

    v8f acc = {};
    for (int kc = 0; kc < KTILES; ++kc) {
        union { v16h v; v8h h[2]; } a;
        int abase = vrow * NDET + kc * 32 + lh * 16;
        a.h[0] = *(const v8h*)&A16[abase];
        a.h[1] = *(const v8h*)&A16[abase + 8];
        v16h b = *(const v16h*)&W16[(size_t)ncol * NDET + kc * 32 + lh * 16];
        acc = __builtin_amdgcn_wmma_f32_16x16x32_f16(false, a.v, false, b,
                                                     (short)0, acc, false, false);
    }
    float bb = bias[ncol];
#pragma unroll
    for (int r = 0; r < 8; ++r) {
        int m  = r + 8 * lh;
        int vv = mt * 16 + m;
        float val = acc[r] + bb;
        val = val > 0.0f ? val : 0.0f;
        if (flags & 1) outH[(size_t)vv * NDET + ncol] = (_Float16)val;
        if ((flags & 2) && vv < VIEWS) outF[(size_t)vv * NDET + ncol] = val;
    }
}

// ---------------------------------------------------------------------------
// Backprojection: one wave per pixel; lanes stride over 360 views.
// img[(511-ix)*512 + (511-iy)] applies flip(axis=(2,3)) at store time.
__global__ __launch_bounds__(256) void backproj_k(const float* __restrict__ sino,
                                                  const int* __restrict__ idx,
                                                  float* __restrict__ img) {
    int wave = (blockIdx.x * 256 + threadIdx.x) >> 5;
    int lane = threadIdx.x & 31;
    const int* ip = idx + (size_t)wave * VIEWS;
    float s = 0.0f;
    for (int v = lane; v < VIEWS; v += 32) s += sino[ip[v]];
#pragma unroll
    for (int off = 16; off; off >>= 1) s += __shfl_xor(s, off, 32);
    if (lane == 0) {
        int ix = wave >> 9, iy = wave & 511;
        img[(size_t)(511 - ix) * 512 + (511 - iy)] = s * BP_SCALE;
    }
}

// ---------------------------------------------------------------------------
// conv_in: 1 -> 64 channels, 3x3 SAME, relu. img f32 [512][512] -> NHWC f16.
__global__ __launch_bounds__(256) void conv_in_k(const float* __restrict__ img,
                                                 const float* __restrict__ w,  // [64][9]
                                                 const float* __restrict__ b,
                                                 _Float16* __restrict__ out) {
    int p = blockIdx.x * 256 + threadIdx.x;
    if (p >= NX * NY) return;
    int y = p >> 9, x = p & 511;
    float pix[9];
#pragma unroll
    for (int ky = 0; ky < 3; ++ky)
#pragma unroll
        for (int kx = 0; kx < 3; ++kx) {
            int yy = y + ky - 1, xx = x + kx - 1;
            pix[ky * 3 + kx] = ((unsigned)yy < 512u && (unsigned)xx < 512u)
                                   ? img[(size_t)yy * 512 + xx] : 0.0f;
        }
#pragma unroll
    for (int c8 = 0; c8 < 8; ++c8) {
        v8h ov;
#pragma unroll
        for (int j = 0; j < 8; ++j) {
            int co = c8 * 8 + j;
            float s = b[co];
#pragma unroll
            for (int t = 0; t < 9; ++t) s += w[co * 9 + t] * pix[t];
            ov[j] = (_Float16)(s > 0.0f ? s : 0.0f);
        }
        *(v8h*)&out[(size_t)p * 64 + c8 * 8] = ov;
    }
}

// ---------------------------------------------------------------------------
// 64->64 3x3 SAME conv via implicit-GEMM WMMA.
// Block = 256 threads (8 waves), computes one row y, 64 x-positions, all 64
// output channels: 16 tiles of (16 co x 16 px), 2 per wave, 18 WMMAs per tile.
// Halo staged into LDS with the CDNA5 async-to-LDS engine when available
// (ASYNCcnt, no VGPR round-trip); zero padding goes through normal DS stores.
// mode 1: relu(conv+b). mode 2: relu(conv+b+res).
__global__ __launch_bounds__(256) void conv3x3_wmma_k(const _Float16* __restrict__ in,
                                                      const _Float16* __restrict__ res,
                                                      _Float16* __restrict__ out,
                                                      const _Float16* __restrict__ wpack,
                                                      const float* __restrict__ bias,
                                                      int mode) {
    __shared__ _Float16 lds[3 * 66 * 64];   // halo tile, NHWC, 25.3 KB
    const int y   = blockIdx.y;
    const int x0  = blockIdx.x * 64;
    const int tid = threadIdx.x;

    // Stage 3 rows x 66 x-positions x 64 f16 channels (zero-padded halo).
    for (int i = tid; i < 3 * 66 * 8; i += 256) {
        int c8 = i & 7;
        int xi = (i >> 3) % 66;
        int r  = (i >> 3) / 66;
        int yy = y - 1 + r;
        int xx = x0 + xi - 1;
        int lidx = (r * 66 + xi) * 64 + c8 * 8;
        if ((unsigned)yy < 512u && (unsigned)xx < 512u) {
            size_t gidx = ((size_t)(yy * 512 + xx)) * 64 + c8 * 8;
#if defined(HAS_ASYNC_LDS)
            __builtin_amdgcn_global_load_async_to_lds_b128(
                (glob_v4i*)&in[gidx],
                (lds_v4i*)&lds[lidx],
                0, 0);
#else
            *(v8h*)&lds[lidx] = *(const v8h*)&in[gidx];
#endif
            if (yy + 1 < 512)   // warm L2 for the next row's halo
                __builtin_prefetch(&in[gidx + 512 * 64], 0, 0);
        } else {
            v8h z = {};
            *(v8h*)&lds[lidx] = z;
        }
    }
#if defined(HAS_ASYNC_LDS)
#if defined(HAS_WAIT_ASYNC)
    __builtin_amdgcn_s_wait_asynccnt(0);
#else
    asm volatile("s_wait_asynccnt 0x0" ::: "memory");
#endif
#endif
    __syncthreads();

    const int lane = tid & 31;
    const int wave = tid >> 5;
    const int lm = lane & 15;
    const int lh = lane >> 4;

    for (int t = wave; t < 16; t += 8) {
        const int co_t = t >> 2;
        const int xt   = t & 3;
        v8f acc = {};
#pragma unroll
        for (int ky = 0; ky < 3; ++ky)
#pragma unroll
            for (int kx = 0; kx < 3; ++kx)
#pragma unroll
                for (int cb = 0; cb < 2; ++cb) {
                    int f = ((co_t * 3 + ky) * 3 + kx) * 2 + cb;
                    v16h a = *(const v16h*)&wpack[((size_t)f * 32 + lane) * 16];
                    int xi = xt * 16 + lm + kx;         // lds x (halo already +1)
                    v16h b = *(const v16h*)&lds[(ky * 66 + xi) * 64 + cb * 32 + lh * 16];
                    acc = __builtin_amdgcn_wmma_f32_16x16x32_f16(false, a, false, b,
                                                                 (short)0, acc, false, false);
                }
        int x   = x0 + xt * 16 + lm;
        int co0 = co_t * 16 + lh * 8;                   // 8 contiguous channels
        size_t obase = ((size_t)(y * 512 + x)) * 64 + co0;
        v8h rv = {};
        if (mode == 2) rv = *(const v8h*)&res[obase];
        v8h ov;
#pragma unroll
        for (int r = 0; r < 8; ++r) {
            float v = acc[r] + bias[co0 + r];
            if (mode == 2) v += (float)rv[r];
            v = v > 0.0f ? v : 0.0f;
            ov[r] = (_Float16)v;
        }
        *(v8h*)&out[obase] = ov;
    }
}

// ---------------------------------------------------------------------------
// conv_out: 64 -> 1, 3x3 SAME, no relu. NHWC f16 -> f32 output.
__global__ __launch_bounds__(256) void conv_out_k(const _Float16* __restrict__ act,
                                                  const float* __restrict__ w,  // [64][9]
                                                  const float* __restrict__ b,
                                                  float* __restrict__ out) {
    int p = blockIdx.x * 256 + threadIdx.x;
    if (p >= NX * NY) return;
    int y = p >> 9, x = p & 511;
    float s = b[0];
    for (int ky = 0; ky < 3; ++ky) {
        int yy = y + ky - 1;
        if ((unsigned)yy >= 512u) continue;
        for (int kx = 0; kx < 3; ++kx) {
            int xx = x + kx - 1;
            if ((unsigned)xx >= 512u) continue;
            const _Float16* ap = &act[((size_t)(yy * 512 + xx)) * 64];
#pragma unroll
            for (int c8 = 0; c8 < 8; ++c8) {
                v8h a = *(const v8h*)&ap[c8 * 8];
#pragma unroll
                for (int j = 0; j < 8; ++j)
                    s += (float)a[j] * w[(c8 * 8 + j) * 9 + ky * 3 + kx];
            }
        }
    }
    out[p] = s;
}

// ---------------------------------------------------------------------------
extern "C" void kernel_launch(void* const* d_in, const int* in_sizes, int n_in,
                              void* d_out, int out_size, void* d_ws, size_t ws_size,
                              hipStream_t stream) {
    const float* x    = (const float*)d_in[0];
    const int*   idx  = (const int*)d_in[1];
    const float* w1   = (const float*)d_in[2];
    const float* b1   = (const float*)d_in[3];
    const float* w2   = (const float*)d_in[4];
    const float* b2   = (const float*)d_in[5];
    const float* ciw  = (const float*)d_in[6];
    const float* cib  = (const float*)d_in[7];
    const float* bw1  = (const float*)d_in[8];
    const float* bb1  = (const float*)d_in[9];
    const float* bw2  = (const float*)d_in[10];
    const float* bb2  = (const float*)d_in[11];
    const float* cow  = (const float*)d_in[12];
    const float* cob  = (const float*)d_in[13];
    float* outp = (float*)d_out;

    char* ws = (char*)d_ws;
    size_t off = 0;
    auto alloc = [&](size_t bytes) -> void* {
        void* p = ws + off;
        off += (bytes + 255) & ~(size_t)255;
        return p;
    };
    _Float16* w1h   = (_Float16*)alloc((size_t)NDET * NDET * 2);
    _Float16* w2h   = (_Float16*)alloc((size_t)NDET * NDET * 2);
    _Float16* x16   = (_Float16*)alloc((size_t)MPAD * NDET * 2);
    _Float16* h16   = (_Float16*)alloc((size_t)MPAD * NDET * 2);
    float*    sino  = (float*)   alloc((size_t)VIEWS * NDET * 4);
    float*    img   = (float*)   alloc((size_t)NX * NY * 4);
    _Float16* wpack = (_Float16*)alloc((size_t)22 * 72 * 32 * 16 * 2);
    _Float16* actA  = (_Float16*)alloc((size_t)NX * NY * 64 * 2);
    _Float16* actB  = (_Float16*)alloc((size_t)NX * NY * 64 * 2);
    (void)ws_size;

    // --- precision conversion + weight packing ---
    {
        int n = NDET * NDET;
        cvt_f16_k<<<(n + 255) / 256, 256, 0, stream>>>(w1, w1h, n, n);
        cvt_f16_k<<<(n + 255) / 256, 256, 0, stream>>>(w2, w2h, n, n);
        int np = MPAD * NDET, ns = VIEWS * NDET;
        cvt_f16_k<<<(np + 255) / 256, 256, 0, stream>>>(x, x16, np, ns);
        pack_conv_w_k<<<22 * 72, 32, 0, stream>>>(bw1, bw2, wpack);
    }

    // --- two dense layers (WMMA) ---
    {
        dim3 g(MPAD / 16, NDET / 16);  // 23 x 46
        gemm_wmma_k<<<g, 32, 0, stream>>>(x16, w1h, b1, h16, sino, 1);
        gemm_wmma_k<<<g, 32, 0, stream>>>(h16, w2h, b2, h16, sino, 2);
    }

    // --- fan-beam backprojection gather (+flip, +scale) ---
    backproj_k<<<(NX * NY) / 8, 256, 0, stream>>>(sino, idx, img);

    // --- CNN ---
    conv_in_k<<<(NX * NY + 255) / 256, 256, 0, stream>>>(img, ciw, cib, actA);

    dim3 cg(NX / 64, NY);  // 8 x 512
    for (int i = 0; i < 11; ++i) {
        const _Float16* wp1 = wpack + (size_t)(2 * i)     * 72 * 32 * 16;
        const _Float16* wp2 = wpack + (size_t)(2 * i + 1) * 72 * 32 * 16;
        conv3x3_wmma_k<<<cg, 256, 0, stream>>>(actA, actA, actB, wp1, bb1 + i * 64, 1);
        conv3x3_wmma_k<<<cg, 256, 0, stream>>>(actB, actA, actA, wp2, bb2 + i * 64, 2);
    }

    conv_out_k<<<(NX * NY + 255) / 256, 256, 0, stream>>>(actA, cow, cob, outp);
}
